// TransformerBlock_65867618451857
// MI455X (gfx1250) — matrix operand order repaired and checked by reading, not compiled, not run on hardware
//
#include <hip/hip_runtime.h>
#include <hip/hip_bf16.h>
#include <math.h>

// ---------------- constants ----------------
#define D_MODEL 1024
#define N_HEADS 16
#define D_HEAD  64
#define D_FF    4096
#define BATCH   2
#define SEQ     2048
#define TOK     (BATCH*SEQ)   // 4096 rows total

typedef __bf16 bf16_t;
typedef __attribute__((ext_vector_type(16))) __bf16 v16bf;
typedef __attribute__((ext_vector_type(8)))  __bf16 v8bf;
typedef __attribute__((ext_vector_type(8)))  float  v8f;
typedef __attribute__((ext_vector_type(4)))  unsigned int u32x4;
typedef __attribute__((ext_vector_type(8)))  int i32x8;
typedef __attribute__((ext_vector_type(4)))  int i32x4;

static __device__ inline v16bf join8(v8bf lo, v8bf hi) {
  return __builtin_shufflevector(lo, hi, 0,1,2,3,4,5,6,7,8,9,10,11,12,13,14,15);
}

static __device__ inline v8f wmma_bf16(v16bf a, v16bf b, v8f c) {
  return __builtin_amdgcn_wmma_f32_16x16x32_bf16(false, a, false, b, (short)0, c, false, false);
}

// async copy 16B global -> LDS (ASYNCcnt path); lds_off is LDS byte offset
static __device__ inline void async_copy_b128(unsigned lds_off, const bf16_t* gptr) {
  unsigned long long ga = (unsigned long long)(uintptr_t)gptr;
  asm volatile("global_load_async_to_lds_b128 %0, %1, off"
               :: "v"(lds_off), "v"(ga) : "memory");
}

// TDM: DMA a 2D tile (tile_d0 x tile_d1 elements, 2-byte elements) from a row-major
// tensor (row length tensor_d0, row stride stride0, tensor_d1 rows) into LDS at lds_off.
// Descriptor packed per CDNA5 ISA ch.8 (count=1, type=2 "image", data_size=1 -> 2B).
static __device__ inline void tdm_load_2d(unsigned lds_off, const void* gaddr,
                                          unsigned tensor_d0, unsigned tensor_d1,
                                          unsigned tile_d0, unsigned tile_d1,
                                          unsigned stride0_elems) {
  const unsigned long long ga = (unsigned long long)(uintptr_t)gaddr;
  u32x4 g0 = { 1u,                                   // count=1, user descriptor
               lds_off,                              // lds_addr (bytes)
               (unsigned)ga,                         // global_addr[31:0]
               (unsigned)((ga >> 32) & 0x01FFFFFFu) | (2u << 30) };  // addr[56:32] | type=2
  i32x8 g1;
  g1[0] = 0x10000;                                              // wg_mask=0, data_size=1 (2B)
  g1[1] = (int)((tensor_d0 & 0xFFFFu) << 16);                   // tensor_dim0[15:0]
  g1[2] = (int)(((tensor_d0 >> 16) & 0xFFFFu) |
                ((tensor_d1 & 0xFFFFu) << 16));                 // dim0[31:16] | dim1[15:0]
  g1[3] = (int)(((tensor_d1 >> 16) & 0xFFFFu) |
                ((tile_d0 & 0xFFFFu) << 16));                   // dim1[31:16] | tile_dim0
  g1[4] = (int)(tile_d1 & 0xFFFFu);                             // tile_dim1 | tile_dim2=0
  g1[5] = (int)stride0_elems;                                   // tensor_dim0_stride[31:0]
  g1[6] = 0;                                                    // stride0[47:32]|stride1[15:0]
  g1[7] = 0;                                                    // stride1[47:16]
  i32x4 z4 = { 0, 0, 0, 0 };
  i32x8 z8 = { 0, 0, 0, 0, 0, 0, 0, 0 };
  __builtin_amdgcn_tensor_load_to_lds(g0, g1, z4, z4, z8, 0);
}

// ---------------- fp32 [K,N] -> bf16 transposed [N,K] ----------------
__global__ __launch_bounds__(256) void cvt_transpose_kernel(
    const float* __restrict__ in, bf16_t* __restrict__ out, int K, int N) {
  __shared__ float t[32][33];
  const int tx = threadIdx.x & 31;
  const int ty = threadIdx.x >> 5;   // 0..7
#pragma unroll
  for (int i = 0; i < 4; i++) {
    const int kk = blockIdx.y * 32 + ty + i * 8;
    t[ty + i * 8][tx] = in[(size_t)kk * N + blockIdx.x * 32 + tx];
  }
  __syncthreads();
#pragma unroll
  for (int i = 0; i < 4; i++) {
    const int nn = blockIdx.x * 32 + ty + i * 8;
    out[(size_t)nn * K + blockIdx.y * 32 + tx] = (bf16_t)t[tx][ty + i * 8];
  }
}

// ---------------- LayerNorm (fp32 in) -> bf16 out ----------------
__global__ __launch_bounds__(256) void ln_bf16_kernel(
    const float* __restrict__ x, const float* __restrict__ g,
    const float* __restrict__ b, bf16_t* __restrict__ out) {
  const int row = blockIdx.x;
  const float* xr = x + (size_t)row * D_MODEL;
  float s = 0.f, s2 = 0.f;
  for (int i = threadIdx.x; i < D_MODEL; i += 256) {
    float v = xr[i]; s += v; s2 += v * v;
  }
  __shared__ float rs[256], rq[256];
  rs[threadIdx.x] = s; rq[threadIdx.x] = s2;
  __syncthreads();
  for (int off = 128; off > 0; off >>= 1) {
    if (threadIdx.x < off) { rs[threadIdx.x] += rs[threadIdx.x + off]; rq[threadIdx.x] += rq[threadIdx.x + off]; }
    __syncthreads();
  }
  const float mu  = rs[0] * (1.0f / D_MODEL);
  const float var = rq[0] * (1.0f / D_MODEL) - mu * mu;
  const float inv = rsqrtf(var + 1e-6f);
  for (int i = threadIdx.x; i < D_MODEL; i += 256)
    out[(size_t)row * D_MODEL + i] = (bf16_t)((xr[i] - mu) * inv * g[i] + b[i]);
}

// ---------------- tiled bf16 WMMA GEMM, double-buffered B staging ----------------
// C[M,N] = A[M,K] @ W + bias where WT is W transposed, [N,K] bf16 (contiguous along K).
// Block tile 128(M) x 64(N), 8 waves; wave = 16 rows x four 16x16 accumulators; K step 32.
// B tile [64 n][32 k] staged to LDS either by the Tensor Data Mover (TENSORcnt) or by
// per-thread GLOBAL_LOAD_ASYNC_TO_LDS_B128 (ASYNCcnt); double buffered in both cases.
// EPI: 0 = bf16 out; 1 = fp32 out + residual; 2 = exact GELU -> bf16;
//      3 = bf16 out transposed per head -> vt[b][col][s] (for attention V operand)
template <int EPI, bool USE_TDM>
__global__ __launch_bounds__(256) void gemm_bf16_kernel(
    const bf16_t* __restrict__ A, const bf16_t* __restrict__ WT,
    const float* __restrict__ bias, const float* __restrict__ resid,
    bf16_t* __restrict__ outH, float* __restrict__ outF,
    int M, int N, int K) {
  __shared__ __align__(16) bf16_t Bs[2][64 * 32];   // [buf][n_local*32 + k_local]
  const int lane = threadIdx.x & 31;
  const int w    = threadIdx.x >> 5;
  const int l15  = lane & 15;
  const int hi   = lane >> 4;
  const int n0   = blockIdx.x * 64;
  const int rowbase = blockIdx.y * 128 + w * 16;
  const int m    = rowbase + l15;
  const int kbA  = hi ? 8 : 0;
  const int kb2  = hi ? 16 : 0;

  // async staging assignment: one 16B chunk per thread per K-step
  const int nl = threadIdx.x >> 2;        // 0..63
  const int ko = (threadIdx.x & 3) * 8;   // 0,8,16,24

  v8f acc[4] = {};

  // prologue: stage first B tile
  if (USE_TDM) {
    if (w == 0)
      tdm_load_2d((unsigned)(uintptr_t)&Bs[0][0], WT + (size_t)n0 * K,
                  (unsigned)K, (unsigned)N, 32u, 64u, (unsigned)K);
  } else {
    async_copy_b128((unsigned)(uintptr_t)&Bs[0][nl * 32 + ko],
                    WT + (size_t)(n0 + nl) * K + ko);
  }

  int cur = 0;
  for (int k0 = 0; k0 < K; k0 += 32, cur ^= 1) {
    const bool has_next = (k0 + 32) < K;
    if (USE_TDM) {
      if (w == 0) {
        if (has_next) {
          tdm_load_2d((unsigned)(uintptr_t)&Bs[cur ^ 1][0],
                      WT + (size_t)n0 * K + (k0 + 32),
                      (unsigned)K, (unsigned)N, 32u, 64u, (unsigned)K);
          __builtin_amdgcn_s_wait_tensorcnt(1);
        } else {
          __builtin_amdgcn_s_wait_tensorcnt(0);
        }
      }
    } else {
      if (has_next) {
        async_copy_b128((unsigned)(uintptr_t)&Bs[cur ^ 1][nl * 32 + ko],
                        WT + (size_t)(n0 + nl) * K + (k0 + 32) + ko);
        asm volatile("s_wait_asynccnt 0x1" ::: "memory");
      } else {
        asm volatile("s_wait_asynccnt 0x0" ::: "memory");
      }
    }
    __syncthreads();   // current tile visible to all waves

    // A fragment (16-bit A 16x32 layout), direct from global
    const bf16_t* ap = A + (size_t)m * K + k0 + kb2;
    const v16bf af = join8(*(const v8bf*)ap, *(const v8bf*)(ap + 8));

#pragma unroll
    for (int j = 0; j < 4; j++) {
      const bf16_t* bp = &Bs[cur][(j * 16 + l15) * 32 + kb2];
      const v16bf bf = join8(*(const v8bf*)bp, *(const v8bf*)(bp + 8));
      acc[j] = wmma_bf16(af, bf, acc[j]);
    }
    __syncthreads();   // everyone done reading before this buffer is re-staged
  }

  // epilogue: C/D layout -> element e of lane: row = e + 8*hi, col = l15
#pragma unroll
  for (int j = 0; j < 4; j++) {
#pragma unroll
    for (int e = 0; e < 8; e++) {
      const int row = rowbase + e + 8 * hi;
      const int col = n0 + j * 16 + l15;
      float val = acc[j][e] + bias[col];
      if (EPI == 0) {
        outH[(size_t)row * N + col] = (bf16_t)val;
      } else if (EPI == 1) {
        const size_t idx = (size_t)row * N + col;
        outF[idx] = resid[idx] + val;
      } else if (EPI == 2) {
        const float gelu = 0.5f * val * (1.0f + erff(val * 0.70710678118654752f));
        outH[(size_t)row * N + col] = (bf16_t)gelu;
      } else { // EPI == 3: per-head transposed store: vt[(b*1024 + col)*SEQ + s]
        const int bb = row >> 11;         // row / SEQ
        const int ss = row & (SEQ - 1);
        outH[((size_t)(bb << 10) + col) * SEQ + ss] = (bf16_t)val;
      }
    }
  }
}

// ---------------- flash attention (bf16 WMMA, fp32 softmax) ----------------
// grid: (SEQ/128, BATCH*N_HEADS), 256 threads; waves fully independent (16 query rows each).
// K fragments contiguous from k[tok][1024]; V fragments contiguous from vt[b][col][s].
__global__ __launch_bounds__(256) void attn_kernel(
    const bf16_t* __restrict__ q, const bf16_t* __restrict__ k,
    const bf16_t* __restrict__ vt, bf16_t* __restrict__ ctx) {
  const int lane = threadIdx.x & 31;
  const int w    = threadIdx.x >> 5;
  const int l15  = lane & 15;
  const int hi   = lane >> 4;
  const int b    = blockIdx.y >> 4;
  const int h    = blockIdx.y & 15;
  const int hoff = h * D_HEAD;
  const int rowbase = b * SEQ + blockIdx.x * 128 + w * 16;
  const int kbA  = hi ? 8 : 0;
  const int kb2  = hi ? 16 : 0;

  __shared__ __align__(16) bf16_t Pt[8][16][40];    // per-wave prob tile: [row][key0..31]

  // Q A-fragments (d 0..31 and d 32..63), loaded once
  const bf16_t* qp = q + (size_t)(rowbase + l15) * D_MODEL + hoff;
  const v16bf qf0 = join8(*(const v8bf*)(qp + kb2),      *(const v8bf*)(qp + 8 + kb2));
  const v16bf qf1 = join8(*(const v8bf*)(qp + 32 + kb2), *(const v8bf*)(qp + 40 + kb2));

  // per-lane base pointers
  const bf16_t* kbase = k + (size_t)(b * SEQ + l15) * D_MODEL + hoff;          // + key*D_MODEL
  const bf16_t* vbase = vt + ((size_t)(b << 10) + hoff + l15) * SEQ;           // + j*16*SEQ + key

  v8f acc[4] = {};
  float mrow[8], lrow[8];
#pragma unroll
  for (int e = 0; e < 8; e++) { mrow[e] = -1e30f; lrow[e] = 0.f; }

  for (int c = 0; c < SEQ / 32; ++c) {
    // scores: keys c*32..+15 (tile0) and +16..+31 (tile1); each = 2 chained WMMAs over d=64
    const bf16_t* kp0 = kbase + (size_t)(c * 32) * D_MODEL;
    const bf16_t* kp1 = kp0 + (size_t)16 * D_MODEL;
    // prefetch next chunk's K rows
    __builtin_prefetch(kp0 + (size_t)32 * D_MODEL, 0, 1);
    __builtin_prefetch(kp1 + (size_t)32 * D_MODEL, 0, 1);

    const v16bf kf00 = join8(*(const v8bf*)(kp0 + kb2),      *(const v8bf*)(kp0 + kb2 + 8));
    const v16bf kf01 = join8(*(const v8bf*)(kp0 + 32 + kb2), *(const v8bf*)(kp0 + 40 + kb2));
    const v16bf kf10 = join8(*(const v8bf*)(kp1 + kb2),      *(const v8bf*)(kp1 + kb2 + 8));
    const v16bf kf11 = join8(*(const v8bf*)(kp1 + 32 + kb2), *(const v8bf*)(kp1 + 40 + kb2));

    v8f s0 = {}, s1 = {};
    s0 = wmma_bf16(qf0, kf00, s0);
    s0 = wmma_bf16(qf1, kf01, s0);
    s1 = wmma_bf16(qf0, kf10, s1);
    s1 = wmma_bf16(qf1, kf11, s1);

    // online softmax: row stats live in 16-lane half-wave groups
    float p0[8], p1[8], alpha[8];
#pragma unroll
    for (int e = 0; e < 8; e++) {
      const float x0 = s0[e] * 0.125f;   // 1/sqrt(64)
      const float x1 = s1[e] * 0.125f;
      float mx = fmaxf(x0, x1);
      mx = fmaxf(mx, __shfl_xor(mx, 1, 16));
      mx = fmaxf(mx, __shfl_xor(mx, 2, 16));
      mx = fmaxf(mx, __shfl_xor(mx, 4, 16));
      mx = fmaxf(mx, __shfl_xor(mx, 8, 16));
      const float mnew = fmaxf(mrow[e], mx);
      alpha[e] = __expf(mrow[e] - mnew);
      mrow[e] = mnew;
      p0[e] = __expf(x0 - mnew);
      p1[e] = __expf(x1 - mnew);
      float rsum = p0[e] + p1[e];
      rsum += __shfl_xor(rsum, 1, 16);
      rsum += __shfl_xor(rsum, 2, 16);
      rsum += __shfl_xor(rsum, 4, 16);
      rsum += __shfl_xor(rsum, 8, 16);
      lrow[e] = lrow[e] * alpha[e] + rsum;
    }
#pragma unroll
    for (int j = 0; j < 4; j++)
#pragma unroll
      for (int e = 0; e < 8; e++) acc[j][e] *= alpha[e];

    // bounce P through per-wave LDS: C-layout -> bf16 A-layout (same-wave LDS ops are in-order)
#pragma unroll
    for (int e = 0; e < 8; e++) {
      Pt[w][e + 8 * hi][l15]      = (bf16_t)p0[e];
      Pt[w][e + 8 * hi][16 + l15] = (bf16_t)p1[e];
    }
    __asm__ volatile("" ::: "memory");

    const bf16_t* pp = &Pt[w][l15][0];
    const v16bf pf = join8(*(const v8bf*)(pp + kb2), *(const v8bf*)(pp + 8 + kb2));

    // PV: V fragments contiguous along keys from vt
#pragma unroll
    for (int j = 0; j < 4; j++) {
      const bf16_t* vp = vbase + (size_t)(j * 16) * SEQ + c * 32 + kb2;
      const v16bf vf = join8(*(const v8bf*)vp, *(const v8bf*)(vp + 8));
      acc[j] = wmma_bf16(pf, vf, acc[j]);
    }
  }

  // normalize and store context (bf16)
#pragma unroll
  for (int j = 0; j < 4; j++)
#pragma unroll
    for (int e = 0; e < 8; e++) {
      const int row = rowbase + e + 8 * hi;
      const int col = hoff + j * 16 + l15;
      ctx[(size_t)row * D_MODEL + col] = (bf16_t)(acc[j][e] / lrow[e]);
    }
}

// ---------------- launch ----------------
extern "C" void kernel_launch(void* const* d_in, const int* in_sizes, int n_in,
                              void* d_out, int out_size, void* d_ws, size_t ws_size,
                              hipStream_t stream) {
  const float* x   = (const float*)d_in[0];
  const float* Wq  = (const float*)d_in[1];  const float* bq  = (const float*)d_in[2];
  const float* Wk  = (const float*)d_in[3];  const float* bk  = (const float*)d_in[4];
  const float* Wv  = (const float*)d_in[5];  const float* bv  = (const float*)d_in[6];
  const float* Wo  = (const float*)d_in[7];  const float* bo  = (const float*)d_in[8];
  const float* W1  = (const float*)d_in[9];  const float* b1  = (const float*)d_in[10];
  const float* W2  = (const float*)d_in[11]; const float* b2  = (const float*)d_in[12];
  const float* g1  = (const float*)d_in[13]; const float* be1 = (const float*)d_in[14];
  const float* g2  = (const float*)d_in[15]; const float* be2 = (const float*)d_in[16];

  char* ws = (char*)d_ws;
  size_t off = 0;
  auto alloc = [&](size_t bytes) -> void* {
    void* p = ws + off;
    off = (off + bytes + 255) & ~(size_t)255;
    return p;
  };

  bf16_t* WqT = (bf16_t*)alloc((size_t)D_MODEL * D_MODEL * 2);
  bf16_t* WkT = (bf16_t*)alloc((size_t)D_MODEL * D_MODEL * 2);
  bf16_t* WvT = (bf16_t*)alloc((size_t)D_MODEL * D_MODEL * 2);
  bf16_t* WoT = (bf16_t*)alloc((size_t)D_MODEL * D_MODEL * 2);
  bf16_t* W1T = (bf16_t*)alloc((size_t)D_MODEL * D_FF * 2);
  bf16_t* W2T = (bf16_t*)alloc((size_t)D_FF * D_MODEL * 2);
  bf16_t* xn1 = (bf16_t*)alloc((size_t)TOK * D_MODEL * 2);
  bf16_t* qb  = (bf16_t*)alloc((size_t)TOK * D_MODEL * 2);
  bf16_t* kb_ = (bf16_t*)alloc((size_t)TOK * D_MODEL * 2);
  bf16_t* vtb = (bf16_t*)alloc((size_t)TOK * D_MODEL * 2);   // V transposed per head
  bf16_t* ctx = (bf16_t*)alloc((size_t)TOK * D_MODEL * 2);
  float*  x1  = (float*) alloc((size_t)TOK * D_MODEL * 4);
  bf16_t* xn2 = (bf16_t*)alloc((size_t)TOK * D_MODEL * 2);
  bf16_t* hh  = (bf16_t*)alloc((size_t)TOK * D_FF * 2);
  (void)ws_size; (void)in_sizes; (void)n_in; (void)out_size;

  // 1) weight conversions: fp32 [K,N] -> bf16 transposed [N,K]
  cvt_transpose_kernel<<<dim3(D_MODEL/32, D_MODEL/32), 256, 0, stream>>>(Wq, WqT, D_MODEL, D_MODEL);
  cvt_transpose_kernel<<<dim3(D_MODEL/32, D_MODEL/32), 256, 0, stream>>>(Wk, WkT, D_MODEL, D_MODEL);
  cvt_transpose_kernel<<<dim3(D_MODEL/32, D_MODEL/32), 256, 0, stream>>>(Wv, WvT, D_MODEL, D_MODEL);
  cvt_transpose_kernel<<<dim3(D_MODEL/32, D_MODEL/32), 256, 0, stream>>>(Wo, WoT, D_MODEL, D_MODEL);
  cvt_transpose_kernel<<<dim3(D_FF/32,    D_MODEL/32), 256, 0, stream>>>(W1, W1T, D_MODEL, D_FF);
  cvt_transpose_kernel<<<dim3(D_MODEL/32, D_FF/32),    256, 0, stream>>>(W2, W2T, D_FF, D_MODEL);

  // 2) pre-norm 1
  ln_bf16_kernel<<<TOK, 256, 0, stream>>>(x, g1, be1, xn1);

  // 3) QKV projections (bf16 WMMA, TDM-staged B); V written per-head transposed
  const dim3 gN1024(D_MODEL / 64, TOK / 128);
  gemm_bf16_kernel<0, true><<<gN1024, 256, 0, stream>>>(xn1, WqT, bq, nullptr, qb,  nullptr, TOK, D_MODEL, D_MODEL);
  gemm_bf16_kernel<0, true><<<gN1024, 256, 0, stream>>>(xn1, WkT, bk, nullptr, kb_, nullptr, TOK, D_MODEL, D_MODEL);
  gemm_bf16_kernel<3, true><<<gN1024, 256, 0, stream>>>(xn1, WvT, bv, nullptr, vtb, nullptr, TOK, D_MODEL, D_MODEL);

  // 4) attention
  attn_kernel<<<dim3(SEQ / 128, BATCH * N_HEADS), 256, 0, stream>>>(qb, kb_, vtb, ctx);

  // 5) output projection + residual -> x1 (fp32), TDM-staged B
  gemm_bf16_kernel<1, true><<<gN1024, 256, 0, stream>>>(ctx, WoT, bo, x, nullptr, x1, TOK, D_MODEL, D_MODEL);

  // 6) pre-norm 2
  ln_bf16_kernel<<<TOK, 256, 0, stream>>>(x1, g2, be2, xn2);

  // 7) FFN up + exact GELU -> bf16, TDM-staged B
  gemm_bf16_kernel<2, true><<<dim3(D_FF / 64, TOK / 128), 256, 0, stream>>>(xn2, W1T, b1, nullptr, hh, nullptr, TOK, D_FF, D_MODEL);

  // 8) FFN down + residual -> d_out (fp32), async-LDS-staged B (ASYNCcnt path)
  gemm_bf16_kernel<1, false><<<gN1024, 256, 0, stream>>>(hh, W2T, b2, x1, nullptr, (float*)d_out, TOK, D_MODEL, D_FF);
}
